// SplitAttentionOp_44229573214696
// MI455X (gfx1250) — matrix operand order repaired and checked by reading, not compiled, not run on hardware
//
#include <hip/hip_runtime.h>

typedef __attribute__((ext_vector_type(16))) _Float16 v16h;
typedef __attribute__((ext_vector_type(8)))  _Float16 v8h;
typedef __attribute__((ext_vector_type(4)))  _Float16 v4h;
typedef __attribute__((ext_vector_type(8)))  float    v8f;

#define SA_H    16
#define SA_SQ   1024
#define SA_SKV  2048
#define SA_D    128
// scale * log2(e): softmax runs in the log2 domain (v_exp_f32 is exp2)
#define SA_SCL2 0.12751242239999f    // (1/sqrt(128)) * 1.4426950408889634
#define SA_LN2  0.6931471805599453f

// LDS pitches in fp16 elements (conflict-free b128 lane reads)
#define KP 136   // K tile: 32 rows x (128+8)
#define VP 40    // V^T tile: 128 rows x (32+8)
#define PP 40    // P scratch: 16 rows x (32+8)
#define NSTEP (2 * SA_SKV / 32)

__device__ __forceinline__ v16h cat16(v8h lo, v8h hi) {
  v16h r;
#pragma unroll
  for (int i = 0; i < 8; ++i) { r[i] = lo[i]; r[i + 8] = hi[i]; }
  return r;
}

// 16-lane row reductions in the VALU via DPP16 (no DS-pipe bpermute traffic)
template <int CTRL>
__device__ __forceinline__ float dpp_f32(float x) {
  return __builtin_bit_cast(float,
      __builtin_amdgcn_update_dpp(0, __builtin_bit_cast(int, x),
                                  CTRL, 0xf, 0xf, true));
}
__device__ __forceinline__ float rowmax16(float x) {
  x = fmaxf(x, dpp_f32<0xB1>(x));    // quad_perm(1,0,3,2)
  x = fmaxf(x, dpp_f32<0x4E>(x));    // quad_perm(2,3,0,1)
  x = fmaxf(x, dpp_f32<0x141>(x));   // row_half_mirror
  x = fmaxf(x, dpp_f32<0x140>(x));   // row_mirror
  return x;
}
__device__ __forceinline__ float rowsum16(float x) {
  x += dpp_f32<0xB1>(x);
  x += dpp_f32<0x4E>(x);
  x += dpp_f32<0x141>(x);
  x += dpp_f32<0x140>(x);
  return x;
}

__global__ __launch_bounds__(128)
void split_attn_fused(const float* __restrict__ q,
                      const float* __restrict__ kA, const float* __restrict__ vA,
                      const float* __restrict__ kB, const float* __restrict__ vB,
                      float* __restrict__ out, float* __restrict__ lse)
{
  __shared__ __align__(16) _Float16 kbuf[32 * KP];        // 8704 B
  __shared__ __align__(16) _Float16 vbufT[128 * VP];      // 10240 B
  __shared__ __align__(16) _Float16 pbuf[4][16 * PP];     // 5120 B

  const int tid  = threadIdx.x;
  const int lane = tid & 31;
  const int wave = tid >> 5;
  const int half = lane >> 4;
  const int l16  = lane & 15;

  const int h    = blockIdx.x % SA_H;
  const int q0   = (blockIdx.x / SA_H) * 64;
  const int qrow = q0 + wave * 16;

  // staging work split (128 threads)
  const int krow = tid >> 2;            // K: one kv row, 32 d columns
  const int kcb  = (tid & 3) * 32;
  const int vkv  = (tid >> 4) * 4;      // V: 4 kv rows x 8 d columns
  const int vdc  = (tid & 15) * 8;

  // ---- Q -> 4 A-fragments (16x32 over d), scale*log2e folded in.
  // A layout (16-bit 16x32): lanes 0-15 hold K = c*32 + {0..7, 16..23},
  // lanes 16-31 hold K = c*32 + {8..15, 24..31}; row M = lane%16.
  v16h qf[4];
  {
    const float* qp = q + (size_t)(h * SA_SQ + qrow + l16) * SA_D;
#pragma unroll
    for (int c = 0; c < 4; ++c) {
      const int dlo = c * 32 + half * 8;
      const int dhi = c * 32 + 16 + half * 8;
      float4 a0 = *(const float4*)(qp + dlo);
      float4 a1 = *(const float4*)(qp + dlo + 4);
      float4 b0 = *(const float4*)(qp + dhi);
      float4 b1 = *(const float4*)(qp + dhi + 4);
      v16h f;
      f[0]  = (_Float16)(a0.x * SA_SCL2); f[1]  = (_Float16)(a0.y * SA_SCL2);
      f[2]  = (_Float16)(a0.z * SA_SCL2); f[3]  = (_Float16)(a0.w * SA_SCL2);
      f[4]  = (_Float16)(a1.x * SA_SCL2); f[5]  = (_Float16)(a1.y * SA_SCL2);
      f[6]  = (_Float16)(a1.z * SA_SCL2); f[7]  = (_Float16)(a1.w * SA_SCL2);
      f[8]  = (_Float16)(b0.x * SA_SCL2); f[9]  = (_Float16)(b0.y * SA_SCL2);
      f[10] = (_Float16)(b0.z * SA_SCL2); f[11] = (_Float16)(b0.w * SA_SCL2);
      f[12] = (_Float16)(b1.x * SA_SCL2); f[13] = (_Float16)(b1.y * SA_SCL2);
      f[14] = (_Float16)(b1.z * SA_SCL2); f[15] = (_Float16)(b1.w * SA_SCL2);
      qf[c] = f;
    }
  }

  v8f o[8];
#pragma unroll
  for (int n = 0; n < 8; ++n)
#pragma unroll
    for (int r = 0; r < 8; ++r) o[n][r] = 0.0f;

  float mrun[8], lrun[8];
#pragma unroll
  for (int r = 0; r < 8; ++r) { mrun[r] = -1.0e30f; lrun[r] = 0.0f; }

  // double-buffer registers for the software-pipelined K/V tile loads
  float kr[32];      // 1 kv row x 32 d
  float vr[4][8];    // 4 kv rows x 8 d

  auto preload = [&](int step) {
    const bool second = step >= SA_SKV / 32;
    const float* ks = second ? kB : kA;
    const float* vs = second ? vB : vA;
    const int j = (second ? step - SA_SKV / 32 : step) * 32;
    const float* kbase = ks + (size_t)(h * SA_SKV + j) * SA_D;
    const float* vbase = vs + (size_t)(h * SA_SKV + j) * SA_D;
    const float* srcK = kbase + krow * SA_D + kcb;
#pragma unroll
    for (int i = 0; i < 8; ++i) {
      float4 x = *(const float4*)(srcK + i * 4);
      kr[i*4+0] = x.x; kr[i*4+1] = x.y; kr[i*4+2] = x.z; kr[i*4+3] = x.w;
    }
#pragma unroll
    for (int rr = 0; rr < 4; ++rr) {
      const float* srcV = vbase + (vkv + rr) * SA_D + vdc;
      float4 a = *(const float4*)(srcV);
      float4 b = *(const float4*)(srcV + 4);
      vr[rr][0] = a.x; vr[rr][1] = a.y; vr[rr][2] = a.z; vr[rr][3] = a.w;
      vr[rr][4] = b.x; vr[rr][5] = b.y; vr[rr][6] = b.z; vr[rr][7] = b.w;
    }
  };

  // K B-fragment load (tile t: kv rows t*16.., chunk c: d = c*32..)
  auto ldK = [&](int c, int t) -> v16h {
    const _Float16* kp = kbuf + (t * 16 + l16) * KP + c * 32 + half * 8;
    return cat16(*(const v8h*)kp, *(const v8h*)(kp + 16));
  };
  // V B-fragment load (d-chunk n)
  auto ldV = [&](int n) -> v16h {
    const _Float16* vp = vbufT + (n * 16 + l16) * VP + half * 8;
    return cat16(*(const v8h*)vp, *(const v8h*)(vp + 16));
  };

  preload(0);

  // One online-softmax pass over the CONCATENATED KV (A then B):
  // exactly equivalent to the reference's LSE-weighted merge.
  for (int step = 0; step < NSTEP; ++step) {
    __syncthreads();   // previous step's LDS consumers are done

    // ---- commit staged K tile (32 x 128) row-major as fp16
    {
      _Float16* dst = kbuf + krow * KP + kcb;
#pragma unroll
      for (int i = 0; i < 4; ++i) {
        v8h p;
#pragma unroll
        for (int e = 0; e < 8; ++e) p[e] = (_Float16)kr[i * 8 + e];
        *(v8h*)(dst + i * 8) = p;
      }
    }
    // ---- commit staged V tile TRANSPOSED (d-major); 4 kv per ds_store_b64
    {
#pragma unroll
      for (int i = 0; i < 8; ++i) {
        v4h w;
        w[0] = (_Float16)vr[0][i]; w[1] = (_Float16)vr[1][i];
        w[2] = (_Float16)vr[2][i]; w[3] = (_Float16)vr[3][i];
        *(v4h*)(vbufT + (vdc + i) * VP + vkv) = w;
      }
    }

    if (step + 1 < NSTEP) preload(step + 1);   // hide HBM/L2 latency

    __syncthreads();

    // ---- S = (Q*scl2) K^T : two 16x16 f32 tiles over 32 kv columns.
    // Double-buffered B fragments so LDS loads overlap the matrix pipe
    // instead of a full s_wait_dscnt drain before every WMMA.
    v8f s0, s1;
#pragma unroll
    for (int r = 0; r < 8; ++r) { s0[r] = 0.0f; s1[r] = 0.0f; }
    {
      v16h b0a = ldK(0, 0), b1a = ldK(0, 1);
      v16h b0b, b1b;
#pragma unroll
      for (int c = 0; c < 4; ++c) {
        v16h& c0 = (c & 1) ? b0b : b0a;
        v16h& c1 = (c & 1) ? b1b : b1a;
        v16h& n0 = (c & 1) ? b0a : b0b;
        v16h& n1 = (c & 1) ? b1a : b1b;
        if (c < 3) { n0 = ldK(c + 1, 0); n1 = ldK(c + 1, 1); }
        s0 = __builtin_amdgcn_wmma_f32_16x16x32_f16(false, qf[c], false, c0,
                                                    (short)0, s0, false, false);
        s1 = __builtin_amdgcn_wmma_f32_16x16x32_f16(false, qf[c], false, c1,
                                                    (short)0, s1, false, false);
      }
    }

    // ---- online softmax in log2 domain; DPP16 row reductions (pure VALU)
    float corr[8];
#pragma unroll
    for (int r = 0; r < 8; ++r) {
      const float mx = rowmax16(fmaxf(s0[r], s1[r]));
      const float mn = fmaxf(mrun[r], mx);
      const float p0 = __builtin_amdgcn_exp2f(s0[r] - mn);
      const float p1 = __builtin_amdgcn_exp2f(s1[r] - mn);
      const float rs = rowsum16(p0 + p1);
      const float cf = __builtin_amdgcn_exp2f(mrun[r] - mn);
      corr[r] = cf;
      lrun[r] = lrun[r] * cf + rs;
      mrun[r] = mn;
      s0[r] = p0; s1[r] = p1;
    }

    // first V fragment: issue early so it overlaps the softmax tail /
    // accumulator rescale (vbufT is stable after the barrier)
    v16h vfa = ldV(0), vfb;

#pragma unroll
    for (int n = 0; n < 8; ++n)
#pragma unroll
      for (int r = 0; r < 8; ++r) o[n][r] *= corr[r];

    // ---- P: accumulator layout -> A-fragment layout via per-wave LDS
    _Float16* pb = pbuf[wave];
#pragma unroll
    for (int r = 0; r < 8; ++r) {
      const int row = r + half * 8;                 // C layout: M = r (+8)
      pb[row * PP + l16]      = (_Float16)s0[r];    // kv cols 0..15
      pb[row * PP + 16 + l16] = (_Float16)s1[r];    // kv cols 16..31
    }
    __asm__ volatile("s_wait_dscnt 0" ::: "memory");  // same-wave cross-lane RAW
    const _Float16* pr = pb + l16 * PP + half * 8;
    v16h pf = cat16(*(const v8h*)pr, *(const v8h*)(pr + 16));

    // ---- O += P V  (8 d-chunks of 16 columns), double-buffered V fragments
#pragma unroll
    for (int n = 0; n < 8; ++n) {
      v16h& cv = (n & 1) ? vfb : vfa;
      v16h& nv = (n & 1) ? vfa : vfb;
      if (n < 7) nv = ldV(n + 1);
      o[n] = __builtin_amdgcn_wmma_f32_16x16x32_f16(false, pf, false, cv,
                                                    (short)0, o[n], false, false);
    }
  }

  // ---- epilogue: out = O / l ; lse = ln2 * (m2 + log2(l))
  float inv[8];
#pragma unroll
  for (int r = 0; r < 8; ++r) inv[r] = 1.0f / lrun[r];
#pragma unroll
  for (int n = 0; n < 8; ++n) {
#pragma unroll
    for (int r = 0; r < 8; ++r) {
      const int row = qrow + r + half * 8;
      out[(size_t)(h * SA_SQ + row) * SA_D + n * 16 + l16] = o[n][r] * inv[r];
    }
  }
  if (l16 == 0) {   // lane 0 writes rows 0..7, lane 16 writes rows 8..15
#pragma unroll
    for (int r = 0; r < 8; ++r) {
      const int row = qrow + r + half * 8;
      lse[h * SA_SQ + row] =
          SA_LN2 * (mrun[r] + __builtin_amdgcn_logf(lrun[r]));
    }
  }
}

extern "C" void kernel_launch(void* const* d_in, const int* in_sizes, int n_in,
                              void* d_out, int out_size, void* d_ws, size_t ws_size,
                              hipStream_t stream) {
  (void)in_sizes; (void)n_in; (void)out_size; (void)d_ws; (void)ws_size;
  const float* q  = (const float*)d_in[0];
  const float* kA = (const float*)d_in[1];
  const float* vA = (const float*)d_in[2];
  const float* kB = (const float*)d_in[3];
  const float* vB = (const float*)d_in[4];
  float* out = (float*)d_out;
  float* lse = out + (size_t)SA_H * SA_SQ * SA_D;   // outputs concatenated
  dim3 grid(SA_H * (SA_SQ / 64));   // 256 blocks: head x 64-row q tile
  dim3 block(128);                  // 4 waves, each owns a 16x128 O tile
  split_attn_fused<<<grid, block, 0, stream>>>(q, kA, vA, kB, vB, out, lse);
}
